// Merge_75084618269189
// MI455X (gfx1250) — matrix operand order repaired and checked by reading, not compiled, not run on hardware
//
#include <hip/hip_runtime.h>
#include <hip/hip_bf16.h>

typedef __attribute__((ext_vector_type(16))) _Float16 v16h;
typedef __attribute__((ext_vector_type(8)))  _Float16 v8h;
typedef __attribute__((ext_vector_type(8)))  float    v8f;

#define BB 8
#define NN 8192
#define CC 32
#define NP 6144
#define HH 4096      // N/2 tokens per side
#define RR 2048      // N - NPOINT merges

union AV { v16h v; struct { v8h lo, hi; } s; };

// ---------------------------------------------------------------------------
// 1) normalize points (f32) -> f16, split even tokens (A) / odd tokens (B)
// ---------------------------------------------------------------------------
__global__ __launch_bounds__(256) void tome_normalize_split(
    const float* __restrict__ points, _Float16* __restrict__ Ah, _Float16* __restrict__ Bh) {
  int t = blockIdx.x * 256 + threadIdx.x;
  if (t >= BB * NN) return;
  int b = t >> 13;
  int n = t & (NN - 1);
  const float* p = points + (size_t)t * CC;
  float v[CC];
  float ss = 0.f;
#pragma unroll
  for (int i = 0; i < CC; i += 4) {
    float4 q = *(const float4*)(p + i);
    v[i] = q.x; v[i + 1] = q.y; v[i + 2] = q.z; v[i + 3] = q.w;
    ss += q.x * q.x + q.y * q.y + q.z * q.z + q.w * q.w;
  }
  float inv = rsqrtf(ss);
  _Float16* dst = ((n & 1) ? Bh : Ah) + ((size_t)b * HH + (n >> 1)) * CC;
#pragma unroll
  for (int i = 0; i < CC; i += 8) {
    v8h h;
#pragma unroll
    for (int j = 0; j < 8; ++j) h[j] = (_Float16)(v[i + j] * inv);
    *(v8h*)(dst + i) = h;
  }
}

// ---------------------------------------------------------------------------
// 2) fused scores GEMM + row max/argmax.
//    One wave owns TWO 16-row M-tiles (32 rows); each B tile feeds 2 WMMAs.
//    N-loop processes 2 B tiles (4 WMMAs) per iteration with CROSS-ITERATION
//    double buffering: next iteration's B tiles are loaded before the current
//    WMMAs, so the s_wait before each WMMA is satisfied ~4 WMMAs + 32 VALU
//    ops in advance (no steady-state load stall).
// ---------------------------------------------------------------------------

// 4 WMMAs + argmax update on a resident pair of B tiles
#define TOME_PROC_PAIR(B0, B1, S0)                                              \
  do {                                                                          \
    v8f cz = {};                                                                \
    v8f c00 = __builtin_amdgcn_wmma_f32_16x16x32_f16(false, a0.v, false, (B0),  \
                                                     (short)0, cz, false, false); \
    v8f c10 = __builtin_amdgcn_wmma_f32_16x16x32_f16(false, a1.v, false, (B0),  \
                                                     (short)0, cz, false, false); \
    v8f c01 = __builtin_amdgcn_wmma_f32_16x16x32_f16(false, a0.v, false, (B1),  \
                                                     (short)0, cz, false, false); \
    v8f c11 = __builtin_amdgcn_wmma_f32_16x16x32_f16(false, a1.v, false, (B1),  \
                                                     (short)0, cz, false, false); \
    int colA = (S0) + l15;                                                      \
    int colB = (S0) + 16 + l15;                                                 \
    _Pragma("unroll")                                                           \
    for (int i = 0; i < 8; ++i) {                                               \
      float v0 = c00[i];                                                        \
      if (v0 > best0[i]) { best0[i] = v0; bidx0[i] = colA; }                    \
      float v1 = c10[i];                                                        \
      if (v1 > best1[i]) { best1[i] = v1; bidx1[i] = colA; }                    \
    }                                                                           \
    _Pragma("unroll")                                                           \
    for (int i = 0; i < 8; ++i) {                                               \
      float v0 = c01[i];                                                        \
      if (v0 > best0[i]) { best0[i] = v0; bidx0[i] = colB; }                    \
      float v1 = c11[i];                                                        \
      if (v1 > best1[i]) { best1[i] = v1; bidx1[i] = colB; }                    \
    }                                                                           \
  } while (0)

__global__ __launch_bounds__(32) void tome_scores_rowmax(
    const _Float16* __restrict__ Ah, const _Float16* __restrict__ Bh,
    float* __restrict__ node_max, int* __restrict__ node_idx) {
  const int blk  = blockIdx.x;
  const int b    = blk >> 7;      // 128 (32-row) M-blocks per batch
  const int mt   = blk & 127;
  const int lane = threadIdx.x;   // wave32
  const int l15  = lane & 15;
  const int hi   = lane >> 4;

  const _Float16* Ab = Ah + (size_t)b * HH * CC;
  const _Float16* Bb = Bh + (size_t)b * HH * CC;

  // A operands, 16x32 f16, lane layout (matches B):
  //  lanes 0-15: row M=l15, K 0-15
  //  lanes 16-31: row M=l15, K 16-31
  const _Float16* arow0 = Ab + (size_t)(mt * 32 + l15) * CC;
  const _Float16* arow1 = arow0 + 16 * CC;
  AV a0, a1;
  a0.s.lo = *(const v8h*)(arow0 + hi * 16);
  a0.s.hi = *(const v8h*)(arow0 + hi * 16 + 8);
  a1.s.lo = *(const v8h*)(arow1 + hi * 16);
  a1.s.hi = *(const v8h*)(arow1 + hi * 16 + 8);

  float best0[8], best1[8];
  int   bidx0[8], bidx1[8];
#pragma unroll
  for (int i = 0; i < 8; ++i) {
    best0[i] = -3.4e38f; bidx0[i] = 0;
    best1[i] = -3.4e38f; bidx1[i] = 0;
  }

  // B operand, K x N = 32 x 16 f16: lane n (0-15) holds column N=n, K 0-15;
  // lane n+16 holds column N=n, K 16-31.  scores = A * B^T so column N=s is
  // b-row s -> 32 contiguous bytes per lane.
  const _Float16* bbase = Bb + (size_t)l15 * CC + hi * 16;

  // prologue: load first pair of B tiles
  v16h b0 = *(const v16h*)(bbase);
  v16h b1 = *(const v16h*)(bbase + (size_t)16 * CC);

  for (int s0 = 0; s0 < HH - 32; s0 += 32) {
    // prefetch next iteration's pair BEFORE consuming the resident one
    v16h n0 = *(const v16h*)(bbase + (size_t)(s0 + 32) * CC);
    v16h n1 = *(const v16h*)(bbase + (size_t)(s0 + 48) * CC);
    TOME_PROC_PAIR(b0, b1, s0);
    b0 = n0;
    b1 = n1;
  }
  // epilogue: last pair (no prefetch)
  TOME_PROC_PAIR(b0, b1, HH - 32);

  // reduce across the 16 lanes of each half (columns), tie-break idx ascending
#pragma unroll
  for (int off = 1; off <= 8; off <<= 1) {
#pragma unroll
    for (int i = 0; i < 8; ++i) {
      float om0 = __shfl_xor(best0[i], off, 32);
      int   oi0 = __shfl_xor(bidx0[i], off, 32);
      if (om0 > best0[i] || (om0 == best0[i] && oi0 < bidx0[i])) {
        best0[i] = om0; bidx0[i] = oi0;
      }
      float om1 = __shfl_xor(best1[i], off, 32);
      int   oi1 = __shfl_xor(bidx1[i], off, 32);
      if (om1 > best1[i] || (om1 == best1[i] && oi1 < bidx1[i])) {
        best1[i] = om1; bidx1[i] = oi1;
      }
    }
  }
  if (l15 == 0) {
#pragma unroll
    for (int i = 0; i < 8; ++i) {
      int row0 = mt * 32 + hi * 8 + i;
      int row1 = row0 + 16;
      node_max[b * HH + row0] = best0[i];
      node_idx[b * HH + row0] = bidx0[i];
      node_max[b * HH + row1] = best1[i];
      node_idx[b * HH + row1] = bidx1[i];
    }
  }
}

// ---------------------------------------------------------------------------
// 3) stable descending argsort of node_max (4096 per batch): bitonic in LDS.
//    Comparator (key desc, idx asc) == stable argsort(-node_max).
// ---------------------------------------------------------------------------
__global__ __launch_bounds__(1024) void tome_argsort(
    const float* __restrict__ node_max, int* __restrict__ edge_idx) {
  __shared__ float key[HH];
  __shared__ int   val[HH];
  const int b = blockIdx.x;
  for (int i = threadIdx.x; i < HH; i += 1024) { key[i] = node_max[b * HH + i]; val[i] = i; }
  __syncthreads();
  for (int k = 2; k <= HH; k <<= 1) {
    for (int j = k >> 1; j > 0; j >>= 1) {
      for (int t = threadIdx.x; t < HH / 2; t += 1024) {
        int i   = ((t & ~(j - 1)) << 1) | (t & (j - 1));
        int ixj = i | j;
        bool up = ((i & k) == 0);
        float ka = key[i], kb = key[ixj];
        int   va = val[i], vb = val[ixj];
        bool aFirst = (ka > kb) || (ka == kb && va < vb);
        if (up ? !aFirst : aFirst) {
          key[i] = kb; key[ixj] = ka;
          val[i] = vb; val[ixj] = va;
        }
      }
      __syncthreads();
    }
  }
  for (int i = threadIdx.x; i < HH; i += 1024) edge_idx[b * HH + i] = val[i];
}

// ---------------------------------------------------------------------------
// 4a) init accumulators: acc = dst rows (odd tokens), cnt = 1
// ---------------------------------------------------------------------------
__global__ __launch_bounds__(256) void tome_init_acc(
    const float* __restrict__ points, const float* __restrict__ xyz,
    float* __restrict__ accP, float* __restrict__ accX, float* __restrict__ cnt) {
  int idx = blockIdx.x * 256 + threadIdx.x;
  if (idx < BB * HH * CC) {
    int b = idx / (HH * CC);
    int rem = idx - b * HH * CC;
    int j = rem >> 5, c = rem & 31;
    accP[idx] = points[((size_t)b * NN + 2 * j + 1) * CC + c];
  }
  if (idx < BB * HH * 3) {
    int b = idx / (HH * 3);
    int rem = idx - b * HH * 3;
    int j = rem / 3, c = rem - 3 * j;
    accX[idx] = xyz[((size_t)b * NN + 2 * j + 1) * 3 + c];
  }
  if (idx < BB * HH) cnt[idx] = 1.0f;
}

// ---------------------------------------------------------------------------
// 4b) scatter-add src rows into their dst accumulator.  One wave per merge.
// ---------------------------------------------------------------------------
__global__ __launch_bounds__(256) void tome_scatter_add(
    const float* __restrict__ points, const float* __restrict__ xyz,
    const int* __restrict__ edge_idx, const int* __restrict__ node_idx,
    float* __restrict__ accP, float* __restrict__ accX, float* __restrict__ cnt) {
  int wid  = (blockIdx.x * 256 + threadIdx.x) >> 5;
  int lane = threadIdx.x & 31;
  if (wid >= BB * RR) return;
  int b = wid >> 11;
  int i = wid & (RR - 1);
  int s = edge_idx[b * HH + i];
  int d = node_idx[b * HH + s];
  const float* prow = points + ((size_t)b * NN + 2 * s) * CC;
  unsafeAtomicAdd(&accP[((size_t)b * HH + d) * CC + lane], prow[lane]);
  if (lane < 3)
    unsafeAtomicAdd(&accX[((size_t)b * HH + d) * 3 + lane],
                    xyz[((size_t)b * NN + 2 * s) * 3 + lane]);
  if (lane == 0) unsafeAtomicAdd(&cnt[b * HH + d], 1.0f);
}

// ---------------------------------------------------------------------------
// 4c) finalize merged points: [unm rows | acc/cnt rows]
// ---------------------------------------------------------------------------
__global__ __launch_bounds__(256) void tome_finalize_points(
    const float* __restrict__ points, const int* __restrict__ edge_idx,
    const float* __restrict__ accP, const float* __restrict__ cnt,
    float* __restrict__ merged) {
  int idx = blockIdx.x * 256 + threadIdx.x;
  if (idx >= BB * NP * CC) return;
  int b = idx / (NP * CC);
  int rem = idx - b * NP * CC;
  int row = rem >> 5, c = rem & 31;
  float out;
  if (row < HH - RR) {  // row < 2048 : unmerged src rows
    int u = edge_idx[b * HH + RR + row];
    out = points[((size_t)b * NN + 2 * u) * CC + c];
  } else {
    int j = row - (HH - RR);
    out = accP[((size_t)b * HH + j) * CC + c] / cnt[b * HH + j];
  }
  merged[idx] = out;
}

__global__ __launch_bounds__(256) void tome_finalize_xyz(
    const float* __restrict__ xyz, const int* __restrict__ edge_idx,
    const float* __restrict__ accX, const float* __restrict__ cnt,
    float* __restrict__ merged_xyz) {
  int idx = blockIdx.x * 256 + threadIdx.x;
  if (idx >= BB * NP * 3) return;
  int b = idx / (NP * 3);
  int rem = idx - b * NP * 3;
  int row = rem / 3, c = rem - 3 * row;
  float out;
  if (row < HH - RR) {
    int u = edge_idx[b * HH + RR + row];
    out = xyz[((size_t)b * NN + 2 * u) * 3 + c];
  } else {
    int j = row - (HH - RR);
    out = accX[((size_t)b * HH + j) * 3 + c] / cnt[b * HH + j];
  }
  merged_xyz[idx] = out;
}

// ---------------------------------------------------------------------------
// 4d) compressed_xyz: odd rows <- merged dst; even rows via edge scatter
// ---------------------------------------------------------------------------
__global__ __launch_bounds__(256) void tome_comp_odd(
    const float* __restrict__ accX, const float* __restrict__ cnt,
    float* __restrict__ comp) {
  int idx = blockIdx.x * 256 + threadIdx.x;
  if (idx >= BB * HH * 3) return;
  int b = idx / (HH * 3);
  int rem = idx - b * HH * 3;
  int j = rem / 3, c = rem - 3 * j;
  comp[((size_t)b * NN + 2 * j + 1) * 3 + c] = accX[idx] / cnt[b * HH + j];
}

__global__ __launch_bounds__(256) void tome_comp_even(
    const float* __restrict__ xyz, const int* __restrict__ edge_idx,
    const int* __restrict__ node_idx, const float* __restrict__ accX,
    const float* __restrict__ cnt, float* __restrict__ comp) {
  int idx = blockIdx.x * 256 + threadIdx.x;
  if (idx >= BB * HH * 3) return;
  int b = idx / (HH * 3);
  int rem = idx - b * HH * 3;
  int i = rem / 3, c = rem - 3 * i;
  int s = edge_idx[b * HH + i];
  float out;
  if (i < RR) {  // merged src position: takes its destination's merged value
    int d = node_idx[b * HH + s];
    out = accX[((size_t)b * HH + d) * 3 + c] / cnt[b * HH + d];
  } else {       // unmerged src position: original xyz
    out = xyz[((size_t)b * NN + 2 * s) * 3 + c];
  }
  comp[((size_t)b * NN + 2 * s) * 3 + c] = out;
}

// ---------------------------------------------------------------------------
extern "C" void kernel_launch(void* const* d_in, const int* in_sizes, int n_in,
                              void* d_out, int out_size, void* d_ws, size_t ws_size,
                              hipStream_t stream) {
  const float* points = (const float*)d_in[0];
  const float* xyz    = (const float*)d_in[1];

  char* ws = (char*)d_ws;
  const size_t szHalf = (size_t)BB * HH * CC * sizeof(_Float16);  // 2 MiB
  const size_t szVec  = (size_t)BB * HH * sizeof(float);          // 128 KiB
  size_t off = 0;
  _Float16* Ah      = (_Float16*)(ws + off); off += szHalf;
  _Float16* Bh      = (_Float16*)(ws + off); off += szHalf;
  float*    nodeMax = (float*)(ws + off);    off += szVec;
  int*      nodeIdx = (int*)(ws + off);      off += szVec;
  int*      edgeIdx = (int*)(ws + off);      off += szVec;
  float*    accP    = (float*)(ws + off);    off += (size_t)BB * HH * CC * sizeof(float);
  float*    accX    = (float*)(ws + off);    off += (size_t)BB * HH * 3 * sizeof(float);
  float*    cnt     = (float*)(ws + off);    off += szVec;

  float* merged     = (float*)d_out;
  float* merged_xyz = merged + (size_t)BB * NP * CC;
  float* comp       = merged_xyz + (size_t)BB * NP * 3;

  tome_normalize_split<<<(BB * NN + 255) / 256, 256, 0, stream>>>(points, Ah, Bh);
  tome_scores_rowmax<<<BB * (HH / 32), 32, 0, stream>>>(Ah, Bh, nodeMax, nodeIdx);
  tome_argsort<<<BB, 1024, 0, stream>>>(nodeMax, edgeIdx);
  tome_init_acc<<<(BB * HH * CC + 255) / 256, 256, 0, stream>>>(points, xyz, accP, accX, cnt);
  tome_scatter_add<<<(BB * RR * 32 + 255) / 256, 256, 0, stream>>>(
      points, xyz, edgeIdx, nodeIdx, accP, accX, cnt);
  tome_finalize_points<<<(BB * NP * CC + 255) / 256, 256, 0, stream>>>(
      points, edgeIdx, accP, cnt, merged);
  tome_finalize_xyz<<<(BB * NP * 3 + 255) / 256, 256, 0, stream>>>(
      xyz, edgeIdx, accX, cnt, merged_xyz);
  tome_comp_odd<<<(BB * HH * 3 + 255) / 256, 256, 0, stream>>>(accX, cnt, comp);
  tome_comp_even<<<(BB * HH * 3 + 255) / 256, 256, 0, stream>>>(
      xyz, edgeIdx, nodeIdx, accX, cnt, comp);
}
